// NystromAttention_19104014532630
// MI455X (gfx1250) — matrix operand order repaired and checked by reading, not compiled, not run on hardware
//
#include <hip/hip_runtime.h>

#define BATCH 4
#define SEQ   4096
#define DIM   768
#define HEADS 12
#define HDIM  64
#define QKV_COLS (3*DIM)
#define MTOT  (BATCH*SEQ)
#define BH    (BATCH*HEADS)

typedef __attribute__((ext_vector_type(16))) __bf16 v16bf;
typedef __attribute__((ext_vector_type(8)))  float  v8f;
typedef __attribute__((ext_vector_type(4)))  float  f32x4;
typedef __attribute__((ext_vector_type(8)))  unsigned short u16x8;
typedef __attribute__((ext_vector_type(4)))  unsigned short u16x4;

union BF16Frag { v16bf v; unsigned short u[16]; u16x8 h[2]; };

__device__ inline unsigned short f2bf(float f) {
  union { __bf16 b; unsigned short u; } c;
  c.b = (__bf16)f;                       // hardware v_cvt_*bf16_f32
  return c.u;
}
__device__ inline float bf2f(unsigned short u) {
  union { unsigned u; float f; } c; c.u = ((unsigned)u) << 16; return c.f;
}

// ---------------- async global->LDS (CDNA5 ASYNCcnt path) --------------------
__device__ inline void async_load_b128(unsigned lds_off, const void* gptr) {
  asm volatile("global_load_async_to_lds_b128 %0, %1, off"
               :: "v"(lds_off), "v"(gptr) : "memory");
}
__device__ inline void wait_async0() {
  asm volatile("s_wait_asynccnt 0" ::: "memory");
}
__device__ inline void wait_async6() {   // <=6 outstanding: previous tile landed
  asm volatile("s_wait_asynccnt 6" ::: "memory");
}

// ---------------- fragment loaders (wave32 WMMA layouts, bf16 sources) -------
// A 16x32: lane m = lane&15; chunks k=[koff..koff+7] and [16+koff..+7] contiguous.
__device__ inline v16bf load_a_frag_w(const unsigned short* A, int lda, int lane) {
  int m    = lane & 15;
  int koff = (lane & 16) ? 8 : 0;
  const unsigned short* base = A + m * lda + koff;
  BF16Frag fr;
  fr.h[0] = *(const u16x8*)(base);
  fr.h[1] = *(const u16x8*)(base + 16);
  return fr.v;
}
// B 32x16 (KxN) from k-contiguous storage Bt[n*ldb + k] (n = B column):
// per lane chunks k=[koff..koff+7] and [16+koff..+7] -> 2 x 16B loads.
__device__ inline v16bf load_bk_frag_w(const unsigned short* Bt, int ldb, int lane) {
  int n  = lane & 15;
  int kh = (lane & 16) ? 8 : 0;
  const unsigned short* base = Bt + n * ldb + kh;
  BF16Frag fr;
  fr.h[0] = *(const u16x8*)(base);
  fr.h[1] = *(const u16x8*)(base + 16);
  return fr.v;
}

// ---------------- wave32 16-lane-half reductions -----------------------------
__device__ inline float half_reduce_max(float x) {
#pragma unroll
  for (int m = 8; m >= 1; m >>= 1) x = fmaxf(x, __shfl_xor(x, m, 32));
  return x;
}
__device__ inline float half_reduce_sum(float x) {
#pragma unroll
  for (int m = 8; m >= 1; m >>= 1) x += __shfl_xor(x, m, 32);
  return x;
}

// =============================================================================
// Prep 1: f32 -> bf16 bulk convert (one-time; x -> xb)
// =============================================================================
__global__ __launch_bounds__(256) void cvt_bf16_kernel(
    const float* __restrict__ X, unsigned short* __restrict__ Xb, int n)
{
  int i = (blockIdx.x * 256 + threadIdx.x) * 8;
  if (i + 8 > n) return;
  f32x4 a = *(const f32x4*)(X + i);
  f32x4 b = *(const f32x4*)(X + i + 4);
  u16x8 o;
  o[0] = f2bf(a[0]); o[1] = f2bf(a[1]); o[2] = f2bf(a[2]); o[3] = f2bf(a[3]);
  o[4] = f2bf(b[0]); o[5] = f2bf(b[1]); o[6] = f2bf(b[2]); o[7] = f2bf(b[3]);
  *(u16x8*)(Xb + i) = o;
}

// =============================================================================
// Prep 2: weight transpose+convert W[K][N] f32 -> Wt[N][K] bf16 (LDS tiled)
// =============================================================================
__global__ __launch_bounds__(256) void transpose_cvt_kernel(
    const float* __restrict__ W, unsigned short* __restrict__ Wt, int K, int N)
{
  __shared__ float tile[64 * 65];   // padded against bank conflicts
  int n0 = blockIdx.x * 64;
  int k0 = blockIdx.y * 64;
  for (int i = threadIdx.x; i < 64 * 16; i += 256) {
    int k = i >> 4, nq = (i & 15) * 4;
    f32x4 v = *(const f32x4*)(W + (k0 + k) * N + n0 + nq);
    tile[k * 65 + nq + 0] = v[0];
    tile[k * 65 + nq + 1] = v[1];
    tile[k * 65 + nq + 2] = v[2];
    tile[k * 65 + nq + 3] = v[3];
  }
  __syncthreads();
  for (int i = threadIdx.x; i < 64 * 16; i += 256) {
    int n = i >> 4, kq = (i & 15) * 4;
    u16x4 o;
    o[0] = f2bf(tile[(kq + 0) * 65 + n]);
    o[1] = f2bf(tile[(kq + 1) * 65 + n]);
    o[2] = f2bf(tile[(kq + 2) * 65 + n]);
    o[3] = f2bf(tile[(kq + 3) * 65 + n]);
    *(u16x4*)(Wt + (n0 + n) * K + k0 + kq) = o;
  }
}

// =============================================================================
// GEMM: acc = Ab[M,K](bf16) @ Bt[N,K](bf16 pre-transposed); K-step 64.
// DOUBLE-BUFFERED async DMA: tile i+1 flies while tile i computes.
// mode 0: C f32 = acc + bias (projection)
// mode 1: QKV epilogue -> qb/kb bf16 row-major, v transposed per head into vt
// =============================================================================
#define ABUF (128 * 64)          // shorts per A tile
#define TILE (ABUF + 64 * 64)    // shorts per (A+B) buffer

__global__ __launch_bounds__(256) void gemm_bf16_kernel(
    const unsigned short* __restrict__ Ab, const unsigned short* __restrict__ Bt,
    const float* __restrict__ bias, float* __restrict__ Cf,
    unsigned short* __restrict__ qb, unsigned short* __restrict__ kb,
    unsigned short* __restrict__ vt,
    int M, int K, int Nn, int mode)
{
  __shared__ alignas(16) unsigned short Sbuf[2 * TILE];   // 48 KB, double buffered

  int tid  = threadIdx.x;
  int lane = tid & 31;
  int wave = tid >> 5;
  int col0 = blockIdx.x * 64;
  int rowb = blockIdx.y * 128;
  int nl   = lane & 15;
  int hi8  = (lane & 16) ? 8 : 0;

  unsigned sbase = (unsigned)(size_t)Sbuf;   // low 32 bits = LDS offset
  int mbase = tid >> 3;                      // 0..31 (row/col group per thread)
  int kq    = (tid & 7) * 8;                 // 8-short chunk within k
  unsigned aoffT = (unsigned)((mbase * 64 + kq) * 2);
  unsigned boffT = (unsigned)(ABUF * 2 + (mbase * 64 + kq) * 2);

  // per-thread async source pointers, advanced by +=64 per K-step
  const unsigned short* paCur = Ab + (rowb + mbase) * K + kq;
  const unsigned short* pbCur = Bt + (col0 + mbase) * K + kq;

  v8f acc[4] = {};
  const int NT = K / 64;

  // prologue: tile 0 -> buffer 0
  {
    unsigned base = sbase;
#pragma unroll
    for (int i = 0; i < 4; ++i)
      async_load_b128(base + aoffT + (unsigned)(i * 4096), paCur + 32 * i * K);
#pragma unroll
    for (int i = 0; i < 2; ++i)
      async_load_b128(base + boffT + (unsigned)(i * 4096), pbCur + 32 * i * K);
  }

  for (int it = 0; it < NT; ++it) {
    if (it + 1 < NT) {
      // issue tile it+1 into the other buffer, then wait for tile it only
      paCur += 64; pbCur += 64;
      unsigned base = sbase + (unsigned)(((it + 1) & 1) * TILE * 2);
#pragma unroll
      for (int i = 0; i < 4; ++i)
        async_load_b128(base + aoffT + (unsigned)(i * 4096), paCur + 32 * i * K);
#pragma unroll
      for (int i = 0; i < 2; ++i)
        async_load_b128(base + boffT + (unsigned)(i * 4096), pbCur + 32 * i * K);
      if (it + 2 < NT) __builtin_prefetch(paCur + 64, 0, 3);   // global_prefetch_b8
      wait_async6();     // async loads complete in order -> tile it landed
    } else {
      wait_async0();
    }
    __syncthreads();     // everyone's tile it visible

    const unsigned short* tb  = Sbuf + (it & 1) * TILE;
    const unsigned short* Aw  = tb + (wave * 16) * 64;
    const unsigned short* Btb = tb + ABUF;
    v16bf af[2];
    af[0] = load_a_frag_w(Aw,      64, lane);
    af[1] = load_a_frag_w(Aw + 32, 64, lane);
    v16bf bfr[8];
#pragma unroll
    for (int t = 0; t < 4; ++t) {
      bfr[t]     = load_bk_frag_w(Btb + (16 * t) * 64,      64, lane);
      bfr[4 + t] = load_bk_frag_w(Btb + (16 * t) * 64 + 32, 64, lane);
    }
#pragma unroll
    for (int ks = 0; ks < 2; ++ks)
#pragma unroll
      for (int t = 0; t < 4; ++t)
        acc[t] = __builtin_amdgcn_wmma_f32_16x16x32_bf16(false, af[ks], false, bfr[ks * 4 + t],
                                                         (short)0, acc[t], false, false);
    __syncthreads();     // buffer it&1 free for tile it+2's DMA
  }

  int row0 = rowb + wave * 16;
  if (mode == 0) {
#pragma unroll
    for (int t = 0; t < 4; ++t) {
      float bv = bias[col0 + 16 * t + nl];
#pragma unroll
      for (int v = 0; v < 8; ++v)
        Cf[(row0 + v + hi8) * Nn + col0 + 16 * t + nl] = acc[t][v] + bv;
    }
  } else {
#pragma unroll
    for (int t = 0; t < 4; ++t) {
      int col  = col0 + 16 * t + nl;
      float bv = bias[col];
#pragma unroll
      for (int v = 0; v < 8; ++v) {
        int   r   = row0 + v + hi8;
        float val = acc[t][v] + bv;
        if (col < DIM) {
          qb[r * DIM + col] = f2bf(val);
        } else if (col < 2 * DIM) {
          kb[r * DIM + col - DIM] = f2bf(val);
        } else {
          int c  = col - 2 * DIM;
          int bb = r >> 12, n = r & (SEQ - 1);
          int hh = c >> 6,  d = c & 63;
          vt[(bb * HEADS + hh) * (HDIM * SEQ) + d * SEQ + n] = f2bf(val);
        }
      }
    }
  }
}

// =============================================================================
// Kernel: per-head flash-style  k3v = softmax(q_l @ k^T / 8) @ v   [64x64]
// All operands bf16; every fragment is 2 contiguous 16B global loads.
// =============================================================================
__global__ __launch_bounds__(128) void landmark_ctx_kernel(
    const unsigned short* __restrict__ qb, const unsigned short* __restrict__ kb,
    const unsigned short* __restrict__ vt, float* __restrict__ k3v)
{
  __shared__ alignas(16) unsigned short Pl[4][16 * 64];

  int lane = threadIdx.x & 31;
  int wave = threadIdx.x >> 5;
  int head = blockIdx.x;
  int bb   = head / HEADS;
  int h    = head % HEADS;
  int nl   = lane & 15;
  int hi8  = (lane & 16) ? 8 : 0;
  int m0   = wave * 16;
  unsigned short* Plw = Pl[wave];

  const unsigned short* qbase = qb + (bb * SEQ + 65 * m0) * DIM + h * HDIM;
  v16bf aq0 = load_a_frag_w(qbase,      65 * DIM, lane);
  v16bf aq1 = load_a_frag_w(qbase + 32, 65 * DIM, lane);

  const unsigned short* vhead = vt + head * (HDIM * SEQ);

  v8f accO[4] = {};
  float rm[8], rl[8];
#pragma unroll
  for (int v = 0; v < 8; ++v) { rm[v] = -1e30f; rl[v] = 0.f; }

  for (int n0 = 0; n0 < SEQ; n0 += 64) {
    v8f accS[4] = {};
#pragma unroll
    for (int ks = 0; ks < 2; ++ks) {
      v16bf a = ks ? aq1 : aq0;
#pragma unroll
      for (int t = 0; t < 4; ++t) {
        v16bf b = load_bk_frag_w(kb + (bb * SEQ + n0 + 16 * t) * DIM + h * HDIM + 32 * ks,
                                 DIM, lane);
        accS[t] = __builtin_amdgcn_wmma_f32_16x16x32_bf16(false, a, false, b,
                                                          (short)0, accS[t], false, false);
      }
    }
#pragma unroll
    for (int v = 0; v < 8; ++v) {
      float tm = -1e30f;
#pragma unroll
      for (int t = 0; t < 4; ++t) { accS[t][v] *= 0.125f; tm = fmaxf(tm, accS[t][v]); }
      tm = half_reduce_max(tm);
      float nm   = fmaxf(rm[v], tm);
      float corr = __expf(rm[v] - nm);
      rm[v] = nm;
      float ts = 0.f;
#pragma unroll
      for (int t = 0; t < 4; ++t) {
        float p = __expf(accS[t][v] - nm);
        ts += p;
        Plw[(v + hi8) * 64 + 16 * t + nl] = f2bf(p);
      }
      ts = half_reduce_sum(ts);
      rl[v] = rl[v] * corr + ts;
#pragma unroll
      for (int t = 0; t < 4; ++t) accO[t][v] *= corr;
    }
#pragma unroll
    for (int ks = 0; ks < 2; ++ks) {
      v16bf ap = load_a_frag_w(Plw + 32 * ks, 64, lane);
#pragma unroll
      for (int t = 0; t < 4; ++t) {
        v16bf bv = load_bk_frag_w(vhead + (16 * t) * SEQ + n0 + 32 * ks, SEQ, lane);
        accO[t] = __builtin_amdgcn_wmma_f32_16x16x32_bf16(false, ap, false, bv,
                                                          (short)0, accO[t], false, false);
      }
    }
  }
#pragma unroll
  for (int t = 0; t < 4; ++t)
#pragma unroll
    for (int v = 0; v < 8; ++v)
      k3v[head * 4096 + (m0 + v + hi8) * 64 + 16 * t + nl] = accO[t][v] / rl[v];
}

// =============================================================================
// Kernel: kernel_2 softmax + Newton-Schulz pinv + W = pinv @ k3v (per head).
// =============================================================================
__device__ inline void mm64(const float* __restrict__ A, const float* __restrict__ B,
                            float* __restrict__ C, int tid)
{
  for (int idx = tid; idx < 4096; idx += 256) {
    int i = idx >> 6, j = idx & 63;
    float s = 0.f;
#pragma unroll 8
    for (int d = 0; d < 64; ++d) s += A[i * 64 + d] * B[d * 64 + j];
    C[idx] = s;
  }
}

__global__ __launch_bounds__(256) void pinv_kernel(
    const unsigned short* __restrict__ qb, const unsigned short* __restrict__ kb,
    const float* __restrict__ k3v, float* __restrict__ Wb)
{
  __shared__ float Sm[5 * 4096 + 132];
  float* LA  = Sm;
  float* LZ  = Sm + 4096;
  float* T1  = Sm + 2 * 4096;
  float* T2  = Sm + 3 * 4096;
  float* T3  = Sm + 4 * 4096;
  float* red = Sm + 5 * 4096;

  int tid  = threadIdx.x;
  int head = blockIdx.x;
  int bb   = head / HEADS;
  int h    = head % HEADS;

  for (int idx = tid; idx < 4096; idx += 256) {
    int i = idx >> 6, d = idx & 63;
    int base = (bb * SEQ + 65 * i) * DIM + h * HDIM + d;
    T1[idx] = bf2f(qb[base]);
    T2[idx] = bf2f(kb[base]);
  }
  __syncthreads();
  for (int idx = tid; idx < 4096; idx += 256) {
    int i = idx >> 6, j = idx & 63;
    float s = 0.f;
#pragma unroll 8
    for (int d = 0; d < 64; ++d) s += T1[i * 64 + d] * T2[j * 64 + d];
    LA[idx] = s * 0.125f;
  }
  __syncthreads();
  if (tid < 64) {
    float mx = -1e30f;
    for (int j = 0; j < 64; ++j) mx = fmaxf(mx, LA[tid * 64 + j]);
    float s = 0.f;
    for (int j = 0; j < 64; ++j) { float e = __expf(LA[tid * 64 + j] - mx); LA[tid * 64 + j] = e; s += e; }
    float inv = 1.f / s;
    for (int j = 0; j < 64; ++j) LA[tid * 64 + j] *= inv;
    LA[tid * 64 + tid] += 1e-6f;
  }
  __syncthreads();
  if (tid < 64) {
    float rs = 0.f, cs = 0.f;
    for (int j = 0; j < 64; ++j) { rs += fabsf(LA[tid * 64 + j]); cs += fabsf(LA[j * 64 + tid]); }
    red[tid] = rs; red[64 + tid] = cs;
  }
  __syncthreads();
  if (tid == 0) {
    float mr = 0.f, mc = 0.f;
    for (int j = 0; j < 64; ++j) { mr = fmaxf(mr, red[j]); mc = fmaxf(mc, red[64 + j]); }
    red[128] = 1.f / (mr * mc);
  }
  __syncthreads();
  float alpha = red[128];
  for (int idx = tid; idx < 4096; idx += 256) {
    int i = idx >> 6, j = idx & 63;
    LZ[idx] = LA[j * 64 + i] * alpha;
  }
  __syncthreads();
  for (int it = 0; it < 10; ++it) {
    mm64(LA, LZ, T1, tid); __syncthreads();
    for (int idx = tid; idx < 4096; idx += 256) {
      int i = idx >> 6, j = idx & 63;
      T2[idx] = ((i == j) ? 7.f : 0.f) - T1[idx];
    }
    __syncthreads();
    mm64(T1, T2, T3, tid); __syncthreads();
    for (int idx = tid; idx < 4096; idx += 256) {
      int i = idx >> 6, j = idx & 63;
      T3[idx] = ((i == j) ? 15.f : 0.f) - T3[idx];
    }
    __syncthreads();
    mm64(T1, T3, T2, tid); __syncthreads();
    for (int idx = tid; idx < 4096; idx += 256) {
      int i = idx >> 6, j = idx & 63;
      T2[idx] = ((i == j) ? 13.f : 0.f) - T2[idx];
    }
    __syncthreads();
    mm64(LZ, T2, T3, tid); __syncthreads();
    for (int idx = tid; idx < 4096; idx += 256) LZ[idx] = 0.25f * T3[idx];
    __syncthreads();
  }
  const float* kv = k3v + head * 4096;
  for (int idx = tid; idx < 4096; idx += 256) {
    int i = idx >> 6, j = idx & 63;
    float s = 0.f;
#pragma unroll 8
    for (int d = 0; d < 64; ++d) s += LZ[i * 64 + d] * kv[d * 64 + j];
    Wb[head * 4096 + idx] = s;
  }
}

// =============================================================================
// Kernel: att = softmax(q @ k_l^T / 8) @ W for all rows; writes att in bf16.
// =============================================================================
__global__ __launch_bounds__(256) void nystrom_out_kernel(
    const unsigned short* __restrict__ qb, const unsigned short* __restrict__ kb,
    const float* __restrict__ Wb, unsigned short* __restrict__ attb)
{
  __shared__ alignas(16) unsigned short KLb[64 * 64];   // [j][d] = k_l rows (bf16 copy)
  __shared__ alignas(16) unsigned short WLt[64 * 64];   // [n][k] = W transposed bf16
  __shared__ alignas(16) unsigned short Pl[8][16 * 64];

  int head = blockIdx.y;
  int bb   = head / HEADS;
  int h    = head % HEADS;
  int lane = threadIdx.x & 31;
  int wave = threadIdx.x >> 5;
  int nl   = lane & 15;
  int hi8  = (lane & 16) ? 8 : 0;

  for (int i = threadIdx.x; i < 64 * 16; i += 256) {
    int j = i >> 4, dq = (i & 15) * 4;
    *(u16x4*)(KLb + j * 64 + dq) =
        *(const u16x4*)(kb + (bb * SEQ + 65 * j) * DIM + h * HDIM + dq);
  }
  for (int idx = threadIdx.x; idx < 4096; idx += 256) {
    int k = idx >> 6, n = idx & 63;            // Wb row-major [k][n]
    WLt[n * 64 + k] = f2bf(Wb[head * 4096 + idx]);
  }
  __syncthreads();

  int row0 = blockIdx.x * 128 + wave * 16;
  const unsigned short* qbase = qb + (bb * SEQ + row0) * DIM + h * HDIM;
  v16bf a0 = load_a_frag_w(qbase,      DIM, lane);
  v16bf a1 = load_a_frag_w(qbase + 32, DIM, lane);

  v8f accS[4] = {};
#pragma unroll
  for (int ks = 0; ks < 2; ++ks) {
    v16bf a = ks ? a1 : a0;
#pragma unroll
    for (int t = 0; t < 4; ++t) {
      v16bf b = load_bk_frag_w(KLb + (16 * t) * 64 + 32 * ks, 64, lane);
      accS[t] = __builtin_amdgcn_wmma_f32_16x16x32_bf16(false, a, false, b,
                                                        (short)0, accS[t], false, false);
    }
  }
  unsigned short* Plw = Pl[wave];
#pragma unroll
  for (int v = 0; v < 8; ++v) {
    float mx = -1e30f;
#pragma unroll
    for (int t = 0; t < 4; ++t) { accS[t][v] *= 0.125f; mx = fmaxf(mx, accS[t][v]); }
    mx = half_reduce_max(mx);
    float p[4]; float s = 0.f;
#pragma unroll
    for (int t = 0; t < 4; ++t) { p[t] = __expf(accS[t][v] - mx); s += p[t]; }
    s = half_reduce_sum(s);
    float inv = 1.f / s;
#pragma unroll
    for (int t = 0; t < 4; ++t) Plw[(v + hi8) * 64 + 16 * t + nl] = f2bf(p[t] * inv);
  }
  v8f accW[4] = {};
#pragma unroll
  for (int ks = 0; ks < 2; ++ks) {
    v16bf ap = load_a_frag_w(Plw + 32 * ks, 64, lane);
#pragma unroll
    for (int t = 0; t < 4; ++t) {
      v16bf bw = load_bk_frag_w(WLt + (16 * t) * 64 + 32 * ks, 64, lane);
      accW[t] = __builtin_amdgcn_wmma_f32_16x16x32_bf16(false, ap, false, bw,
                                                        (short)0, accW[t], false, false);
    }
  }
#pragma unroll
  for (int t = 0; t < 4; ++t)
#pragma unroll
    for (int v = 0; v < 8; ++v)
      attb[(bb * SEQ + row0 + v + hi8) * DIM + h * HDIM + 16 * t + nl] = f2bf(accW[t][v]);
}

// =============================================================================
extern "C" void kernel_launch(void* const* d_in, const int* in_sizes, int n_in,
                              void* d_out, int out_size, void* d_ws, size_t ws_size,
                              hipStream_t stream)
{
  (void)in_sizes; (void)n_in; (void)out_size; (void)ws_size;
  const float* x      = (const float*)d_in[0];
  const float* w_qkv  = (const float*)d_in[1];
  const float* b_qkv  = (const float*)d_in[2];
  const float* w_proj = (const float*)d_in[3];
  const float* b_proj = (const float*)d_in[4];
  float* out = (float*)d_out;

  const size_t ACT_BF  = (size_t)MTOT * DIM * 2;       // 25.2 MB each
  const size_t WQKV_BF = (size_t)DIM * QKV_COLS * 2;   // 3.5 MB
  const size_t WPRJ_BF = (size_t)DIM * DIM * 2;        // 1.2 MB
  const size_t HMAT_F  = (size_t)BH * 64 * 64 * 4;     // 768 KB

  char* ws = (char*)d_ws;
  unsigned short* xb    = (unsigned short*)(ws);
  unsigned short* qbuf  = (unsigned short*)(ws + ACT_BF);
  unsigned short* kbuf  = (unsigned short*)(ws + 2 * ACT_BF);
  unsigned short* vtb   = (unsigned short*)(ws + 3 * ACT_BF);
  unsigned short* attb  = (unsigned short*)(ws + 4 * ACT_BF);
  unsigned short* wqkvT = (unsigned short*)(ws + 5 * ACT_BF);
  unsigned short* wprjT = (unsigned short*)(ws + 5 * ACT_BF + WQKV_BF);
  float*          k3v   = (float*)(ws + 5 * ACT_BF + WQKV_BF + WPRJ_BF);
  float*          Wb    = (float*)(ws + 5 * ACT_BF + WQKV_BF + WPRJ_BF + HMAT_F);

  cvt_bf16_kernel<<<dim3((MTOT * DIM) / (256 * 8)), 256, 0, stream>>>(x, xb, MTOT * DIM);
  transpose_cvt_kernel<<<dim3(QKV_COLS / 64, DIM / 64), 256, 0, stream>>>(w_qkv, wqkvT, DIM, QKV_COLS);
  transpose_cvt_kernel<<<dim3(DIM / 64, DIM / 64), 256, 0, stream>>>(w_proj, wprjT, DIM, DIM);

  gemm_bf16_kernel<<<dim3(QKV_COLS / 64, MTOT / 128), 256, 0, stream>>>(
      xb, wqkvT, b_qkv, nullptr, qbuf, kbuf, vtb, MTOT, DIM, QKV_COLS, 1);
  landmark_ctx_kernel<<<dim3(BH), 128, 0, stream>>>(qbuf, kbuf, vtb, k3v);
  pinv_kernel<<<dim3(BH), 256, 0, stream>>>(qbuf, kbuf, k3v, Wb);
  nystrom_out_kernel<<<dim3(SEQ / 128, BH), 256, 0, stream>>>(qbuf, kbuf, Wb, attb);
  gemm_bf16_kernel<<<dim3(DIM / 64, MTOT / 128), 256, 0, stream>>>(
      attb, wprjT, b_proj, out, nullptr, nullptr, nullptr, MTOT, DIM, DIM, 0);
}
